// EnsembleGILRLayer_67980742362013
// MI455X (gfx1250) — matrix operand order repaired and checked by reading, not compiled, not run on hardware
//
#include <hip/hip_runtime.h>

// ---------------------------------------------------------------------------
// EnsembleGILRLayer on MI455X (gfx1250): bf16 WMMA GEMMs + chunked fp32 scan
// + joint LayerNorm.  E=8, B=8, T=1024, DIN=D=256. M=B*T=8192, N=K=256.
// ---------------------------------------------------------------------------

typedef __attribute__((ext_vector_type(16))) __bf16         v16bf;
typedef __attribute__((ext_vector_type(8)))  float          v8f;
typedef __attribute__((ext_vector_type(8)))  unsigned short ushort8;

#define M_TOT 8192
#define N_TOT 256
#define K_TOT 256
#define NCH   16384   // E*B*D independent scan channels
#define NCHK  16      // scan chunks (T=1024 -> 16 x 64)
#define CHKL  64

__device__ __forceinline__ unsigned short f2bf(float f) {
  unsigned int u = __float_as_uint(f);
  unsigned int r = u + 0x7FFFu + ((u >> 16) & 1u);   // round-to-nearest-even
  return (unsigned short)(r >> 16);
}
__device__ __forceinline__ float bf2f(unsigned short s) {
  return __uint_as_float(((unsigned int)s) << 16);
}

union FragBF { ushort8 h[2]; v16bf v; };

// ---------------------------------------------------------------------------
// fp32 -> bf16 conversion (grid-stride)
// ---------------------------------------------------------------------------
__global__ void cvt_bf16(const float* __restrict__ src,
                         unsigned short* __restrict__ dst, int n) {
  int i = blockIdx.x * blockDim.x + threadIdx.x;
  int stride = gridDim.x * blockDim.x;
  for (; i < n; i += stride) dst[i] = f2bf(src[i]);
}

// fp32 W [batch][K][N] -> bf16 Wt [batch][N][K]  (one-time, ~5 MB total)
__global__ void cvt_bf16_t(const float* __restrict__ src,
                           unsigned short* __restrict__ dst, int n) {
  int i = blockIdx.x * blockDim.x + threadIdx.x;
  int stride = gridDim.x * blockDim.x;
  for (; i < n; i += stride) {
    int batch = i >> 16;            // / (K*N)
    int rem   = i & 0xFFFF;
    int nn    = rem >> 8;           // / K
    int kk    = rem & 255;
    dst[i] = f2bf(src[((size_t)batch << 16) + ((size_t)kk << 8) + nn]);
  }
}

// ---------------------------------------------------------------------------
// Tiled WMMA GEMM: C[batch] = act(A[batch] (MxK, row-maj) * Wt[batch] (NxK) + b)
// Block tile 128x128, BK=32, 256 threads = 8 waves (wave32), wave tile 32x64.
// MODE: 0 tanh->bf16 | 1 sigmoid*(1-start)->bf16 | 2 linear->bf16
//       3 gelu(exact)->bf16 | 4 bias+residual->fp32
// ---------------------------------------------------------------------------
template <int MODE>
__global__ __launch_bounds__(256) void gemm_bf16(
    const unsigned short* __restrict__ Abase, long aStride,
    const unsigned short* __restrict__ Wtbase,   // [batch][N][K] bf16
    const float* __restrict__ biasBase,
    unsigned short* __restrict__ outB,           // bf16 output (modes 0..3)
    float* __restrict__ outF,                    // fp32 output (mode 4)
    const float* __restrict__ rnnStart,          // mode 1
    const unsigned short* __restrict__ resid) {  // mode 4 (bf16 residual)
  constexpr int BK  = 32;
  constexpr int LDA = 40;  // 80 B row stride (16B multiple, staggered banks)
  constexpr int LDB = 40;

  __shared__ __align__(16) unsigned short As[128 * LDA];  // [m][k]
  __shared__ __align__(16) unsigned short Bs[128 * LDB];  // [n][k]

  const int tid   = threadIdx.x;
  const int lane  = tid & 31;
  const int wave  = tid >> 5;
  const int wm    = wave & 3;   // wave row 0..3  (32 rows each)
  const int wn    = wave >> 2;  // wave col 0..1  (64 cols each)
  const int halfl = lane >> 4;  // lane half (ISA A/B fragment split)
  const int lidx  = lane & 15;

  const int tileM = (blockIdx.x & 63) * 128;
  const int tileN = (blockIdx.x >> 6) * 128;
  const int batch = blockIdx.y;

  const unsigned short* A  = Abase + (size_t)batch * (size_t)aStride;
  const unsigned short* Wt = Wtbase + (size_t)batch * (size_t)(K_TOT * N_TOT);
  const float* bias = biasBase + batch * N_TOT;

  v8f acc[2][4];
#pragma unroll
  for (int i = 0; i < 2; ++i)
#pragma unroll
    for (int j = 0; j < 4; ++j) acc[i][j] = v8f{};

  for (int kk = 0; kk < K_TOT; kk += BK) {
    // ---- stage A tile 128x32 (row-major), 2 x ushort8 per thread ----
#pragma unroll
    for (int it = 0; it < 2; ++it) {
      int idx = tid + it * 256;                  // 0..511
      int r   = idx >> 2;                        // 0..127
      int cs  = (idx & 3) * 8;                   // 0,8,16,24
      ushort8 d = *(const ushort8*)(A + (size_t)(tileM + r) * K_TOT + kk + cs);
      *(ushort8*)(&As[r * LDA + cs]) = d;
    }
    // ---- stage B tile [n=tileN..+127][k=kk..+31] from pre-transposed Wt ----
#pragma unroll
    for (int it = 0; it < 2; ++it) {
      int idx = tid + it * 256;                  // 0..511
      int r   = idx >> 2;                        // 0..127 (n within tile)
      int cs  = (idx & 3) * 8;                   // 0,8,16,24 (k within tile)
      ushort8 d =
          *(const ushort8*)(Wt + (size_t)(tileN + r) * K_TOT + kk + cs);
      *(ushort8*)(&Bs[r * LDB + cs]) = d;
    }
    __syncthreads();

    // near-cache prefetch of next K slab (global_prefetch_b8)
    if (kk + BK < K_TOT) {
      __builtin_prefetch(A + (size_t)(tileM + (tid >> 1)) * K_TOT + kk + BK, 0, 3);
      __builtin_prefetch(Wt + (size_t)(tileN + (tid >> 1)) * K_TOT + kk + BK, 0, 3);
    }

    // ---- A fragments: 16x32 bf16, per ISA lanes 0-15 K[0..7],[16..23],
    //      lanes 16-31 K[8..15],[24..31] for row m = lane%16 ----
    FragBF af[2];
#pragma unroll
    for (int im = 0; im < 2; ++im) {
      int m = wm * 32 + im * 16 + lidx;
      const unsigned short* p = &As[m * LDA + halfl * 8];
      af[im].h[0] = *(const ushort8*)p;          // K = halfl*8 + 0..7
      af[im].h[1] = *(const ushort8*)(p + 16);   // K = 16 + halfl*8 + 0..7
    }
    // ---- B fragments: per-lane column n = lane%16, K = halfl*8+0..7, 16+halfl*8+0..7 ----
    FragBF bfr[4];
#pragma unroll
    for (int in = 0; in < 4; ++in) {
      int n = wn * 64 + in * 16 + lidx;
      const unsigned short* p = &Bs[n * LDB + halfl * 8];
      bfr[in].h[0] = *(const ushort8*)p;
      bfr[in].h[1] = *(const ushort8*)(p + 16);
    }

#pragma unroll
    for (int im = 0; im < 2; ++im)
#pragma unroll
      for (int in = 0; in < 4; ++in)
        acc[im][in] = __builtin_amdgcn_wmma_f32_16x16x32_bf16(
            false, af[im].v, false, bfr[in].v, (short)0, acc[im][in], false,
            false);
    __syncthreads();
  }

  // ---- epilogue: C/D layout — VGPR r, lanes 0-15 -> M=r, lanes 16-31 -> M=r+8,
  //      N = lane%16 ----
#pragma unroll
  for (int im = 0; im < 2; ++im) {
#pragma unroll
    for (int in = 0; in < 4; ++in) {
      int col = tileN + wn * 64 + in * 16 + lidx;
      float bcol = bias[col];
#pragma unroll
      for (int r = 0; r < 8; ++r) {
        int row = tileM + wm * 32 + im * 16 + halfl * 8 + r;
        float u = acc[im][in][r] + bcol;
        size_t oidx = ((size_t)batch * M_TOT + row) * N_TOT + col;
        if (MODE == 0) {
          outB[oidx] = f2bf(tanhf(u));
        } else if (MODE == 1) {
          float s = 1.0f / (1.0f + __expf(-u));
          s *= (1.0f - rnnStart[row]);
          outB[oidx] = f2bf(s);
        } else if (MODE == 2) {
          outB[oidx] = f2bf(u);
        } else if (MODE == 3) {
          outB[oidx] = f2bf(0.5f * u * (1.0f + erff(u * 0.70710678f)));
        } else {  // MODE 4: + residual, fp32 out (pre-LayerNorm h2)
          outF[oidx] = u + bf2f(resid[oidx]);
        }
      }
    }
  }
}

// ---------------------------------------------------------------------------
// Chunked tied-gate scan.  h_t = f_t*h_{t-1} + (1-f_t)*v_t is affine in h, so
// split T=1024 into 16 chunks of 64 to trade a little extra traffic for 16x
// more parallelism (262144 threads in passes 1/3 instead of 16384).
// ---------------------------------------------------------------------------
__device__ __forceinline__ size_t scan_base(int ch, int q) {
  int e = ch >> 11;        // / (B*D)
  int b = (ch >> 8) & 7;
  int d = ch & 255;
  return ((size_t)e * M_TOT + (size_t)b * 1024 + (size_t)q * CHKL) * N_TOT + d;
}

// Pass 1: per-chunk affine composition  h_out = Ac*h_in + Bc
__global__ __launch_bounds__(256) void scan_pass1(
    const unsigned short* __restrict__ vbf,
    const unsigned short* __restrict__ fbf,
    float* __restrict__ Ac, float* __restrict__ Bc) {
  int idx = blockIdx.x * 256 + threadIdx.x;  // 0..262143
  int ch = idx & (NCH - 1);
  int q  = idx >> 14;
  size_t base = scan_base(ch, q);
  float a = 1.0f, bacc = 0.0f;
  for (int t = 0; t < CHKL; ++t) {
    size_t i = base + (size_t)t * N_TOT;
    float f = bf2f(fbf[i]);
    float v = bf2f(vbf[i]);
    bacc = f * bacc + (1.0f - f) * v;
    a *= f;
  }
  Ac[(size_t)q * NCH + ch] = a;
  Bc[(size_t)q * NCH + ch] = bacc;
}

// Pass 2: scan 16 chunk coefficients per channel; emit chunk-entry states and
// the final hidden state.
__global__ __launch_bounds__(256) void scan_pass2(
    const float* __restrict__ Ac, const float* __restrict__ Bc,
    float* __restrict__ Hin, float* __restrict__ hlast) {
  int ch = blockIdx.x * 256 + threadIdx.x;  // 0..16383
  float h = 0.0f;
#pragma unroll
  for (int q = 0; q < NCHK; ++q) {
    Hin[(size_t)q * NCH + ch] = h;
    h = Ac[(size_t)q * NCH + ch] * h + Bc[(size_t)q * NCH + ch];
  }
  hlast[ch] = h;  // hidden [1, E*B, D]: ch = (e*8+b)*256+d  (matches layout)
}

// Pass 3: replay each chunk from its entry state, write seq (bf16)
__global__ __launch_bounds__(256) void scan_pass3(
    const unsigned short* __restrict__ vbf,
    const unsigned short* __restrict__ fbf,
    const float* __restrict__ Hin, unsigned short* __restrict__ seqbf) {
  int idx = blockIdx.x * 256 + threadIdx.x;
  int ch = idx & (NCH - 1);
  int q  = idx >> 14;
  size_t base = scan_base(ch, q);
  float h = Hin[(size_t)q * NCH + ch];
  for (int t = 0; t < CHKL; ++t) {
    size_t i = base + (size_t)t * N_TOT;
    float f = bf2f(fbf[i]);
    float v = bf2f(vbf[i]);
    h = f * h + (1.0f - f) * v;
    seqbf[i] = f2bf(h);
  }
}

// ---------------------------------------------------------------------------
// Joint LayerNorm over (E, D) = 2048 elements per (b,t); in place in d_out.
// ---------------------------------------------------------------------------
__global__ __launch_bounds__(256) void ln_kernel(
    float* __restrict__ h2, const float* __restrict__ scale,
    const float* __restrict__ bias) {
  __shared__ float red[256];
  __shared__ float red2[256];
  int row = blockIdx.x;   // 0..8191
  int d   = threadIdx.x;  // 0..255
  float x[8];
  float s = 0.0f, s2 = 0.0f;
#pragma unroll
  for (int e = 0; e < 8; ++e) {
    x[e] = h2[(size_t)e * (size_t)(M_TOT * N_TOT) + (size_t)row * N_TOT + d];
    s += x[e];
    s2 += x[e] * x[e];
  }
  red[d] = s;
  red2[d] = s2;
  __syncthreads();
  for (int off = 128; off > 0; off >>= 1) {
    if (d < off) {
      red[d] += red[d + off];
      red2[d] += red2[d + off];
    }
    __syncthreads();
  }
  float mu  = red[0] * (1.0f / 2048.0f);
  float var = red2[0] * (1.0f / 2048.0f) - mu * mu;
  float rs  = rsqrtf(var + 1e-5f);
#pragma unroll
  for (int e = 0; e < 8; ++e) {
    float y = (x[e] - mu) * rs * scale[e * 256 + d] + bias[e * 256 + d];
    h2[(size_t)e * (size_t)(M_TOT * N_TOT) + (size_t)row * N_TOT + d] = y;
  }
}

// ---------------------------------------------------------------------------
// Host launch
// ---------------------------------------------------------------------------
extern "C" void kernel_launch(void* const* d_in, const int* in_sizes, int n_in,
                              void* d_out, int out_size, void* d_ws,
                              size_t ws_size, hipStream_t stream) {
  (void)in_sizes; (void)n_in; (void)out_size; (void)ws_size;

  const float* x     = (const float*)d_in[0];   // [B,T,DIN]
  const float* rnn   = (const float*)d_in[1];   // [B,T,1]
  const float* w_in  = (const float*)d_in[2];   // [2,E,DIN,D]
  const float* b_in  = (const float*)d_in[3];   // [2,E,D]
  const float* w_out = (const float*)d_in[4];   // [E,D,D]
  const float* b_out = (const float*)d_in[5];   // [E,D]
  const float* ff_w1 = (const float*)d_in[6];
  const float* ff_b1 = (const float*)d_in[7];
  const float* ff_w2 = (const float*)d_in[8];
  const float* ff_b2 = (const float*)d_in[9];
  const float* ln_s  = (const float*)d_in[10];
  const float* ln_b  = (const float*)d_in[11];

  float* out    = (float*)d_out;                        // [E,B,T,D] fp32
  float* hidden = out + (size_t)8 * M_TOT * N_TOT;      // [1, E*B, D]

  // ---- carve workspace (256B aligned) ----
  char* ws = (char*)d_ws;
  auto carve = [&](size_t bytes) {
    void* p = (void*)ws;
    ws += (bytes + 255) & ~(size_t)255;
    return p;
  };
  const size_t EMN = (size_t)8 * M_TOT * N_TOT;  // 16.7M elems per big tensor
  unsigned short* xbf    = (unsigned short*)carve((size_t)M_TOT * K_TOT * 2);
  unsigned short* winbf  = (unsigned short*)carve((size_t)2 * 8 * K_TOT * N_TOT * 2);
  unsigned short* woutbf = (unsigned short*)carve((size_t)8 * K_TOT * N_TOT * 2);
  unsigned short* w1bf   = (unsigned short*)carve((size_t)8 * K_TOT * N_TOT * 2);
  unsigned short* w2bf   = (unsigned short*)carve((size_t)8 * K_TOT * N_TOT * 2);
  unsigned short* vbf    = (unsigned short*)carve(EMN * 2);
  unsigned short* fbf    = (unsigned short*)carve(EMN * 2);
  unsigned short* seqbf  = (unsigned short*)carve(EMN * 2);
  unsigned short* outbf  = (unsigned short*)carve(EMN * 2);
  unsigned short* h1bf   = (unsigned short*)carve(EMN * 2);
  float* Ac  = (float*)carve((size_t)NCHK * NCH * 4);
  float* Bc  = (float*)carve((size_t)NCHK * NCH * 4);
  float* Hin = (float*)carve((size_t)NCHK * NCH * 4);

  // ---- 1. downconvert x; downconvert + transpose weights (W[K][N]->Wt[N][K])
  cvt_bf16<<<1024, 256, 0, stream>>>(x, xbf, M_TOT * K_TOT);
  cvt_bf16_t<<<1024, 256, 0, stream>>>(w_in, winbf, 2 * 8 * K_TOT * N_TOT);
  cvt_bf16_t<<<512, 256, 0, stream>>>(w_out, woutbf, 8 * K_TOT * N_TOT);
  cvt_bf16_t<<<512, 256, 0, stream>>>(ff_w1, w1bf, 8 * K_TOT * N_TOT);
  cvt_bf16_t<<<512, 256, 0, stream>>>(ff_w2, w2bf, 8 * K_TOT * N_TOT);

  dim3 grid(128, 8);  // 64 M-tiles * 2 N-tiles, 8 ensembles
  const long AS = (long)M_TOT * K_TOT;

  // ---- 2. in_proj: value (tanh) and gate (sigmoid * (1-start)) ----
  gemm_bf16<0><<<grid, 256, 0, stream>>>(xbf, 0, winbf, b_in, vbf, nullptr,
                                         nullptr, nullptr);
  gemm_bf16<1><<<grid, 256, 0, stream>>>(xbf, 0, winbf + 8 * K_TOT * N_TOT,
                                         b_in + 8 * 256, fbf, nullptr, rnn,
                                         nullptr);

  // ---- 3. chunked tied-gate scan (pass2 writes hidden to d_out tail) ----
  scan_pass1<<<1024, 256, 0, stream>>>(vbf, fbf, Ac, Bc);
  scan_pass2<<<64, 256, 0, stream>>>(Ac, Bc, Hin, hidden);
  scan_pass3<<<1024, 256, 0, stream>>>(vbf, fbf, Hin, seqbf);

  // ---- 4. out_proj ----
  gemm_bf16<2><<<grid, 256, 0, stream>>>(seqbf, AS, woutbf, b_out, outbf,
                                         nullptr, nullptr, nullptr);
  // ---- 5. ff1 + exact gelu ----
  gemm_bf16<3><<<grid, 256, 0, stream>>>(outbf, AS, w1bf, ff_b1, h1bf, nullptr,
                                         nullptr, nullptr);
  // ---- 6. ff2 + residual -> fp32 h2 directly into d_out ----
  gemm_bf16<4><<<grid, 256, 0, stream>>>(h1bf, AS, w2bf, ff_b2, nullptr, out,
                                         nullptr, outbf);

  // ---- 7. joint LayerNorm over (E,D), in place ----
  ln_kernel<<<8192, 256, 0, stream>>>(out, ln_s, ln_b);
}
